// MultiLayerGCN_variate_2078764171900
// MI455X (gfx1250) — matrix operand order repaired and checked by reading, not compiled, not run on hardware
//
#include <hip/hip_runtime.h>
#include <hip/hip_bf16.h>

// ---------------------------------------------------------------------------
// MI455X (gfx1250) implementation.
// All GEMM-shaped math goes through v_wmma_f32_16x16x32_f16 (wave32, 16x16
// tiles, K=32). f32 accumulate, f16 operand conversion on the fly.
// ---------------------------------------------------------------------------

#define BB   16
#define MM   512
#define LSEQ 1024
#define DD   128
#define KK   16
#define HH   8
#define DFF  512
#define NLAY 2

typedef _Float16 v16h __attribute__((ext_vector_type(16)));
typedef float    v8f  __attribute__((ext_vector_type(8)));

// ---------------------------------------------------------------------------
// Generic batched WMMA GEMM:  C[b] = op(A[b] * B[b] (+bias)), one wave per
// 16(M) x 64(N) output strip; A fragment reused across 4 WMMAs per k-step.
// Requires M%16==0, N%64==0, K%32==0 (true for every call site here).
// ---------------------------------------------------------------------------
template<bool TRANSB, bool BIAS, bool RELU>
__global__ __launch_bounds__(32)
void wmma_gemm(const float* __restrict__ A, int lda, long sA,
               const float* __restrict__ Bm, int ldb, long sB,
               const float* __restrict__ bias,
               float* __restrict__ C, int ldc, long sC,
               int Mr, int Nc, int Kd)
{
    __shared__ float As[16 * 32];   // 2 KB
    __shared__ float Bs[32 * 64];   // 8 KB
    const int lane   = threadIdx.x;
    const int ntiles = Nc >> 6;
    const int mt     = blockIdx.x / ntiles;
    const int nt4    = blockIdx.x % ntiles;
    const int m0     = mt << 4;
    const int n0     = nt4 << 6;
    A  += (long)blockIdx.y * sA;
    Bm += (long)blockIdx.y * sB;
    C  += (long)blockIdx.y * sC;

    v8f acc[4] = {};
    for (int k0 = 0; k0 < Kd; k0 += 32) {
        // stage A tile (16x32) coalesced, float4 granularity
        #pragma unroll
        for (int t = 0; t < 4; ++t) {
            int c   = lane + 32 * t;          // 0..127
            int row = c >> 3;
            int kk  = (c & 7) << 2;
            float4 v = *(const float4*)(A + (long)(m0 + row) * lda + k0 + kk);
            *(float4*)(&As[row * 32 + kk]) = v;
        }
        // stage B tile (32x64)
        if (TRANSB) {
            #pragma unroll
            for (int t = 0; t < 16; ++t) {
                int c  = lane + 32 * t;       // 0..511
                int n  = c >> 3;
                int kk = (c & 7) << 2;
                float4 v = *(const float4*)(Bm + (long)(n0 + n) * ldb + k0 + kk);
                Bs[(kk + 0) * 64 + n] = v.x;
                Bs[(kk + 1) * 64 + n] = v.y;
                Bs[(kk + 2) * 64 + n] = v.z;
                Bs[(kk + 3) * 64 + n] = v.w;
            }
        } else {
            #pragma unroll
            for (int t = 0; t < 16; ++t) {
                int c  = lane + 32 * t;
                int kk = c >> 4;
                int n  = (c & 15) << 2;
                float4 v = *(const float4*)(Bm + (long)(k0 + kk) * ldb + n0 + n);
                *(float4*)(&Bs[kk * 64 + n]) = v;
            }
        }
        __syncthreads();

        // A fragment: 16-bit A-matrix 16x32 layout (ISA 7.12.2)
        v16h a;
        const int rowA = lane & 15;
        #pragma unroll
        for (int j = 0; j < 16; ++j) {
            int kk = ((j & 8) << 1) + ((lane >> 4) << 3) + (j & 7);
            a[j] = (_Float16)As[rowA * 32 + kk];
        }
        #pragma unroll
        for (int nt = 0; nt < 4; ++nt) {
            v16h bf;
            const int n = (nt << 4) + (lane & 15);
            #pragma unroll
            for (int j = 0; j < 16; ++j) {
                int kk = ((j & 8) << 1) + ((lane >> 4) << 3) + (j & 7);   // B 32x16 layout
                bf[j] = (_Float16)Bs[kk * 64 + n];
            }
            acc[nt] = __builtin_amdgcn_wmma_f32_16x16x32_f16(
                false, a, false, bf, (short)0, acc[nt], false, false);
        }
        __syncthreads();
    }

    #pragma unroll
    for (int nt = 0; nt < 4; ++nt) {
        const int ncol = n0 + (nt << 4) + (lane & 15);
        #pragma unroll
        for (int r = 0; r < 8; ++r) {
            const int mrow = m0 + r + ((lane >> 4) << 3);
            float v = acc[nt][r];
            if (BIAS) v += bias[ncol];
            if (RELU) v = fmaxf(v, 0.f);
            C[(long)mrow * ldc + ncol] = v;
        }
    }
}

// ---------------------------------------------------------------------------
// Row mean + inverse centered-norm of x_enc^T rows (coalesced over m).
// ---------------------------------------------------------------------------
__global__ void rowstats_kernel(const float* __restrict__ xe,
                                float* __restrict__ mu, float* __restrict__ rn)
{
    int idx = blockIdx.x * blockDim.x + threadIdx.x;   // b*M + m
    if (idx >= BB * MM) return;
    int b = idx >> 9, m = idx & (MM - 1);
    const float* p = xe + (long)b * LSEQ * MM + m;
    float s = 0.f, sq = 0.f;
    for (int l = 0; l < LSEQ; ++l) {
        float v = p[(long)l * MM];
        s += v; sq += v * v;
    }
    float mean = s * (1.f / LSEQ);
    float c2   = sq - (float)LSEQ * mean * mean;   // sum((x-mu)^2)
    mu[idx] = mean;
    rn[idx] = (c2 > 0.f) ? rsqrtf(c2) : 0.f;       // (L-1) cancels in corr
}

// Y[b,m,l] = (x_enc[b,l,m]-mu)/||c||  (transpose absorbed by 192MB L2)
__global__ void buildY_kernel(const float* __restrict__ xe,
                              const float* __restrict__ mu,
                              const float* __restrict__ rn,
                              float* __restrict__ Y)
{
    long t = (long)blockIdx.x * blockDim.x + threadIdx.x;  // b*M*L + m*L + l
    int  l = (int)(t & (LSEQ - 1));
    long r = t >> 10;                                      // b*M + m
    int  m = (int)(r & (MM - 1));
    int  b = (int)(r >> 9);
    float v = xe[((long)b * LSEQ + l) * MM + m];
    Y[t] = (v - mu[r]) * rn[r];
}

// ---------------------------------------------------------------------------
// Stable ascending top-(K+1) selection per corr row; keep indices 1..K.
// One wave per row, 16 candidates per lane, masked argmin with smaller-index
// tie break (== stable argsort prefix).
// ---------------------------------------------------------------------------
__global__ __launch_bounds__(32)
void topk_kernel(const float* __restrict__ corr, int* __restrict__ neigh)
{
    const int row  = blockIdx.x;          // b*M + m
    const int lane = threadIdx.x;
    const float* p = corr + (long)row * MM;
    float vals[16];
    #pragma unroll
    for (int i = 0; i < 16; ++i) vals[i] = p[lane * 16 + i];
    unsigned mask = 0u;
    for (int it = 0; it <= KK; ++it) {
        float bv = 3.4e38f;
        int   bi = lane * 16;
        #pragma unroll
        for (int i = 0; i < 16; ++i)
            if (!((mask >> i) & 1u) && vals[i] < bv) { bv = vals[i]; bi = lane * 16 + i; }
        #pragma unroll
        for (int off = 16; off; off >>= 1) {
            float ov = __shfl_xor(bv, off, 32);
            int   oi = __shfl_xor(bi, off, 32);
            if (ov < bv || (ov == bv && oi < bi)) { bv = ov; bi = oi; }
        }
        if ((bi >> 4) == lane) mask |= 1u << (bi & 15);
        if (it >= 1 && lane == 0) neigh[(long)row * KK + (it - 1)] = bi;
    }
}

// ---------------------------------------------------------------------------
// Degrees / aggregation (edge list: row tiled mod M, col node-major = faithful
// to reference's repeat/reshape mismatch).
// ---------------------------------------------------------------------------
__global__ void fill_kernel(float* p, int n, float v)
{
    int t = blockIdx.x * blockDim.x + threadIdx.x;
    if (t < n) p[t] = v;
}

__global__ void degree_kernel(const int* __restrict__ neigh, float* __restrict__ deg)
{
    int e = blockIdx.x * blockDim.x + threadIdx.x;
    if (e >= BB * MM * KK) return;
    int b = e / (MM * KK);
    int i = e % (MM * KK);
    int c = neigh[(long)b * MM * KK + i] + b * MM;
    atomicAdd(deg + c, 1.0f);    // integer-valued f32 adds: exact, order-free
}

__global__ void rsqrt_kernel(const float* __restrict__ a, float* __restrict__ o, int n)
{
    int t = blockIdx.x * blockDim.x + threadIdx.x;
    if (t < n) o[t] = rsqrtf(a[t]);   // deg >= 1 always (self loop)
}

// out[n,:] = bias + dinv[n]^2 * xw[n,:]   (self-loop term)
__global__ void agg_init_kernel(const float* __restrict__ xw,
                                const float* __restrict__ dinv,
                                const float* __restrict__ bias,
                                float* __restrict__ out)
{
    int t = blockIdx.x * blockDim.x + threadIdx.x;   // n*D + d
    if (t >= BB * MM * DD) return;
    int n = t >> 7, d = t & (DD - 1);
    float di = dinv[n];
    out[t] = bias[d] + di * di * xw[t];
}

// one wave per edge: out[c,:] += dinv[r]*dinv[c] * xw[r,:]
__global__ __launch_bounds__(256)
void agg_edge_kernel(const int* __restrict__ neigh, const float* __restrict__ dinv,
                     const float* __restrict__ xw, float* __restrict__ out)
{
    int wid  = (blockIdx.x * blockDim.x + threadIdx.x) >> 5;
    int lane = threadIdx.x & 31;
    if (wid >= BB * MM * KK) return;
    int b = wid / (MM * KK);
    int i = wid % (MM * KK);
    int r = b * MM + (i % MM);                       // row: tiled arange
    int c = b * MM + neigh[(long)b * MM * KK + i];   // col: node-major
    float norm = dinv[r] * dinv[c];
    float4 x4 = *(const float4*)(xw + (long)r * DD + lane * 4);
    float* op = out + (long)c * DD + lane * 4;
    atomicAdd(op + 0, norm * x4.x);
    atomicAdd(op + 1, norm * x4.y);
    atomicAdd(op + 2, norm * x4.z);
    atomicAdd(op + 3, norm * x4.w);
}

__global__ void relu_kernel(float* p, int n)
{
    int t = blockIdx.x * blockDim.x + threadIdx.x;
    if (t < n) p[t] = fmaxf(p[t], 0.f);
}

// ---------------------------------------------------------------------------
// Fused cross-attention: one wave per (b, h, 16-query tile). dh=16 padded to
// K=32 for WMMA, with *branchless* padding:
//  - A fragment: valid iff j<8 (compile-time) -> literal zeros for j>=8
//  - B fragment: also valid iff j<8 (compile-time) -> literal zeros for j>=8
// so no exec-mask divergence and head vectors load as float4 (b128).
// Scores (16x512 f32 = 32KB) live in LDS; softmax wave-local; attn*V is a
// K=512 WMMA chain. No [B,H,M,M] HBM traffic.
// ---------------------------------------------------------------------------
__global__ __launch_bounds__(32)
void attn_kernel(const float* __restrict__ Q, const float* __restrict__ Kv,
                 const float* __restrict__ Vv, float* __restrict__ O)
{
    __shared__ float sc[16 * MM];     // 32 KB
    const int lane = threadIdx.x;
    const int mt = blockIdx.x & 31;
    const int h  = (blockIdx.x >> 5) & 7;
    const int b  = blockIdx.x >> 8;
    const int m0 = mt << 4;
    const int dh = DD / HH;           // 16
    const int row = lane & 15;
    const int hi  = lane >> 4;        // 0/1: which half of the K=32 pad

    // Q fragment, scaled by 1/sqrt(dh)=0.25.  kk = (j&8)*2 + hi*8 + (j&7):
    // kk<16 iff j<8, and then kk = hi*8 + j (8 contiguous floats per lane).
    v16h a;
    {
        const float* qrow = Q + ((long)b * MM + m0 + row) * DD + h * dh + hi * 8;
        float4 q0 = *(const float4*)(qrow);
        float4 q1 = *(const float4*)(qrow + 4);
        a[0] = (_Float16)(q0.x * 0.25f);  a[1] = (_Float16)(q0.y * 0.25f);
        a[2] = (_Float16)(q0.z * 0.25f);  a[3] = (_Float16)(q0.w * 0.25f);
        a[4] = (_Float16)(q1.x * 0.25f);  a[5] = (_Float16)(q1.y * 0.25f);
        a[6] = (_Float16)(q1.z * 0.25f);  a[7] = (_Float16)(q1.w * 0.25f);
        #pragma unroll
        for (int j = 8; j < 16; ++j) a[j] = (_Float16)0.f;   // K=16..31 pad
    }

    // scores = Q K^T / sqrt(dh):  B frag kk = (j&8)*2 + hi*8 + (j&7), valid iff j<8.
    for (int kt = 0; kt < MM / 16; ++kt) {
        const int key = (kt << 4) + row;
        const float* krow = Kv + ((long)b * MM + key) * DD + h * dh + hi * 8;
        float4 k0 = *(const float4*)(krow);
        float4 k1 = *(const float4*)(krow + 4);
        v16h bf;
        bf[0]  = (_Float16)(k0.x);  bf[1]  = (_Float16)(k0.y);
        bf[2]  = (_Float16)(k0.z);  bf[3]  = (_Float16)(k0.w);
        bf[4]  = (_Float16)(k1.x);  bf[5]  = (_Float16)(k1.y);
        bf[6]  = (_Float16)(k1.z);  bf[7]  = (_Float16)(k1.w);
        #pragma unroll
        for (int j = 8; j < 16; ++j) bf[j] = (_Float16)0.f;   // K=16..31 pad
        v8f s = {};
        s = __builtin_amdgcn_wmma_f32_16x16x32_f16(false, a, false, bf, (short)0, s, false, false);
        #pragma unroll
        for (int r = 0; r < 8; ++r)
            sc[(r + hi * 8) * MM + key] = s[r];
    }
    __syncthreads();

    // softmax: 2 lanes per row (256 cols each), combined via shfl_xor(16)
    {
        const int base = hi * 256;
        float mx = -3.4e38f;
        for (int i = 0; i < 256; ++i) mx = fmaxf(mx, sc[row * MM + base + i]);
        mx = fmaxf(mx, __shfl_xor(mx, 16, 32));
        float sum = 0.f;
        for (int i = 0; i < 256; ++i) {
            float e = __expf(sc[row * MM + base + i] - mx);
            sc[row * MM + base + i] = e;
            sum += e;
        }
        sum += __shfl_xor(sum, 16, 32);
        float inv = 1.f / sum;
        for (int i = 0; i < 256; ++i) sc[row * MM + base + i] *= inv;
    }
    __syncthreads();

    // O = attn * V  (K = 512, 16 WMMA steps); all indices affine, no pads.
    v8f acc = {};
    for (int kt = 0; kt < MM / 32; ++kt) {
        v16h pa, vb;
        #pragma unroll
        for (int j = 0; j < 8; ++j) {
            pa[j]     = (_Float16)sc[row * MM + (kt << 5) + hi * 8 + j];
            pa[j + 8] = (_Float16)sc[row * MM + (kt << 5) + 16 + hi * 8 + j];
        }
        #pragma unroll
        for (int j = 0; j < 16; ++j) {
            int key = (kt << 5) + ((j & 8) << 1) + hi * 8 + (j & 7);
            vb[j] = (_Float16)Vv[((long)b * MM + key) * DD + h * dh + row];
        }
        acc = __builtin_amdgcn_wmma_f32_16x16x32_f16(false, pa, false, vb, (short)0, acc, false, false);
    }
    #pragma unroll
    for (int r = 0; r < 8; ++r) {
        int mrow = m0 + r + hi * 8;
        O[((long)b * MM + mrow) * DD + h * dh + row] = acc[r];
    }
}

// ---------------------------------------------------------------------------
// out = layernorm(X0 + Y0) * g + b   -- one wave per 128-wide row
// ---------------------------------------------------------------------------
__global__ __launch_bounds__(32)
void addln_kernel(const float* __restrict__ X0, const float* __restrict__ Y0,
                  const float* __restrict__ g, const float* __restrict__ bb,
                  float* __restrict__ out)
{
    const int row  = blockIdx.x;
    const int lane = threadIdx.x;
    float v[4], s = 0.f, sq = 0.f;
    #pragma unroll
    for (int j = 0; j < 4; ++j) {
        int c = lane * 4 + j;
        float x = X0[(long)row * DD + c] + Y0[(long)row * DD + c];
        v[j] = x; s += x; sq += x * x;
    }
    #pragma unroll
    for (int off = 16; off; off >>= 1) {
        s  += __shfl_xor(s,  off, 32);
        sq += __shfl_xor(sq, off, 32);
    }
    float mu  = s * (1.f / DD);
    float var = sq * (1.f / DD) - mu * mu;
    float inv = rsqrtf(var + 1e-5f);
    #pragma unroll
    for (int j = 0; j < 4; ++j) {
        int c = lane * 4 + j;
        out[(long)row * DD + c] = (v[j] - mu) * inv * g[c] + bb[c];
    }
}

// ---------------------------------------------------------------------------
extern "C" void kernel_launch(void* const* d_in, const int* in_sizes, int n_in,
                              void* d_out, int out_size, void* d_ws, size_t ws_size,
                              hipStream_t stream)
{
    const float* enc  = (const float*)d_in[0];
    const float* xe   = (const float*)d_in[1];
    const float* gw1  = (const float*)d_in[2];
    const float* gb1  = (const float*)d_in[3];
    const float* gw2  = (const float*)d_in[4];
    const float* gb2  = (const float*)d_in[5];
    const float* Wq   = (const float*)d_in[6];
    const float* Wk   = (const float*)d_in[7];
    const float* Wv   = (const float*)d_in[8];
    const float* Wo   = (const float*)d_in[9];
    const float* ln1g = (const float*)d_in[10];
    const float* ln1b = (const float*)d_in[11];
    const float* W1   = (const float*)d_in[12];
    const float* b1   = (const float*)d_in[13];
    const float* W2   = (const float*)d_in[14];
    const float* b2   = (const float*)d_in[15];
    const float* ln2g = (const float*)d_in[16];
    const float* ln2b = (const float*)d_in[17];

    char* w = (char*)d_ws;
    const long NROW = (long)BB * MM;              // 8192
    const long SZD  = NROW * DD * 4;              // 4 MiB per [N,128] buffer

    // Phase-1 pool (Y + corr), reused in phase 2 (lifetimes disjoint):
    float* Y    = (float*)(w + 0);                 // 33.5 MB [B,M,L]
    float* corr = (float*)(w + 33554432);          // 16.8 MB [B,M,M]
    float* q    = (float*)(w + 0);
    float* kq   = (float*)(w + 1 * SZD);
    float* vq   = (float*)(w + 2 * SZD);
    float* o    = (float*)(w + 3 * SZD);
    float* o2   = (float*)(w + 4 * SZD);
    float* Xb   = (float*)(w + 5 * SZD);
    float* hb   = (float*)(w + 6 * SZD);
    float* ff1  = (float*)(w + 7 * SZD);           // 16.8 MB [N,512]
    // Persistent region:
    char* pb    = w + 50331648;
    float* mu   = (float*)(pb);
    float* rn   = (float*)(pb + 32768);
    int*   neigh= (int*)  (pb + 65536);            // 512 KB
    float* deg  = (float*)(pb + 589824);
    float* dinv = (float*)(pb + 622592);
    float* x1   = (float*)(pb + 655360);           // 4 MB
    float* kvb  = (float*)(pb + 655360 + SZD);     // 4 MB
    float* xw   = q;                               // alias (dead before q live)

    // ---- Phase 1: correlation graph ----
    rowstats_kernel<<<(BB * MM) / 256, 256, 0, stream>>>(xe, mu, rn);
    buildY_kernel<<<(int)((long)BB * MM * LSEQ / 256), 256, 0, stream>>>(xe, mu, rn, Y);
    wmma_gemm<true, false, false><<<dim3((MM/16)*(MM/64), BB), 32, 0, stream>>>(
        Y, LSEQ, (long)MM * LSEQ, Y, LSEQ, (long)MM * LSEQ, nullptr,
        corr, MM, (long)MM * MM, MM, MM, LSEQ);
    topk_kernel<<<BB * MM, 32, 0, stream>>>(corr, neigh);

    fill_kernel<<<(BB * MM) / 256, 256, 0, stream>>>(deg, BB * MM, 1.0f);   // self loops
    degree_kernel<<<(BB * MM * KK) / 256, 256, 0, stream>>>(neigh, deg);
    rsqrt_kernel<<<(BB * MM) / 256, 256, 0, stream>>>(deg, dinv, BB * MM);

    // ---- Phase 2: 2-layer GCN ----
    const int gemmDD = (int)(NROW / 16) * (DD / 64);          // 1024 blocks
    wmma_gemm<false, false, false><<<dim3(gemmDD, 1), 32, 0, stream>>>(
        enc, DD, 0, gw1, DD, 0, nullptr, xw, DD, 0, (int)NROW, DD, DD);
    agg_init_kernel<<<(int)(NROW * DD / 256), 256, 0, stream>>>(xw, dinv, gb1, x1);
    agg_edge_kernel<<<(BB * MM * KK) / 8, 256, 0, stream>>>(neigh, dinv, xw, x1);
    relu_kernel<<<(int)(NROW * DD / 256), 256, 0, stream>>>(x1, (int)(NROW * DD));

    wmma_gemm<false, false, false><<<dim3(gemmDD, 1), 32, 0, stream>>>(
        x1, DD, 0, gw2, DD, 0, nullptr, xw, DD, 0, (int)NROW, DD, DD);
    agg_init_kernel<<<(int)(NROW * DD / 256), 256, 0, stream>>>(xw, dinv, gb2, kvb);
    agg_edge_kernel<<<(BB * MM * KK) / 8, 256, 0, stream>>>(neigh, dinv, xw, kvb);
    relu_kernel<<<(int)(NROW * DD / 256), 256, 0, stream>>>(kvb, (int)(NROW * DD));

    // ---- Phase 3: transformer encoder (cross-attn to GCN features) ----
    const float* hin = enc;
    for (int l = 0; l < NLAY; ++l) {
        const float* wq = Wq + (long)l * DD * DD;
        const float* wk = Wk + (long)l * DD * DD;
        const float* wv = Wv + (long)l * DD * DD;
        const float* wo = Wo + (long)l * DD * DD;
        const float* w1 = W1 + (long)l * DD * DFF;
        const float* w2 = W2 + (long)l * DFF * DD;

        wmma_gemm<false, false, false><<<dim3(gemmDD, 1), 32, 0, stream>>>(
            hin, DD, 0, wq, DD, 0, nullptr, q, DD, 0, (int)NROW, DD, DD);
        wmma_gemm<false, false, false><<<dim3(gemmDD, 1), 32, 0, stream>>>(
            kvb, DD, 0, wk, DD, 0, nullptr, kq, DD, 0, (int)NROW, DD, DD);
        wmma_gemm<false, false, false><<<dim3(gemmDD, 1), 32, 0, stream>>>(
            kvb, DD, 0, wv, DD, 0, nullptr, vq, DD, 0, (int)NROW, DD, DD);

        attn_kernel<<<BB * HH * (MM / 16), 32, 0, stream>>>(q, kq, vq, o);

        wmma_gemm<false, false, false><<<dim3(gemmDD, 1), 32, 0, stream>>>(
            o, DD, 0, wo, DD, 0, nullptr, o2, DD, 0, (int)NROW, DD, DD);
        addln_kernel<<<(int)NROW, 32, 0, stream>>>(hin, o2, ln1g + l * DD, ln1b + l * DD, Xb);

        wmma_gemm<false, true, true><<<dim3((int)(NROW/16)*(DFF/64), 1), 32, 0, stream>>>(
            Xb, DD, 0, w1, DFF, 0, b1 + l * DFF, ff1, DFF, 0, (int)NROW, DFF, DD);
        wmma_gemm<false, true, false><<<dim3(gemmDD, 1), 32, 0, stream>>>(
            ff1, DFF, 0, w2, DD, 0, b2 + l * DD, o, DD, 0, (int)NROW, DD, DFF);

        float* outp = (l == NLAY - 1) ? (float*)d_out : hb;
        addln_kernel<<<(int)NROW, 32, 0, stream>>>(Xb, o, ln2g + l * DD, ln2b + l * DD, outp);
        hin = hb;
    }
}
